// TripleViewNet_74775380624006
// MI455X (gfx1250) — matrix operand order repaired and checked by reading, not compiled, not run on hardware
//
#include <hip/hip_runtime.h>
#include <hip/hip_bf16.h>
#include <cstdint>

#define HC      64      // H*C per-layer output width
#define NHEAD   2
#define CPH     32
#define FDIM    128
#define JKDIM   192
#define NB      256
#define NEG_SLOPE 0.2f
#define LN_EPS  1e-5f

typedef __attribute__((ext_vector_type(16))) _Float16 v16h;
typedef __attribute__((ext_vector_type(8)))  float    v8f;

// ---------------------------------------------------------------- utilities

static __device__ __forceinline__ void atomicMaxF(float* a, float v) {
    // IEEE-754 ordered-compare trick: non-negative floats compare as signed
    // ints; negative floats compare reversed as unsigned ints.
    if (v >= 0.0f) atomicMax((int*)a, __float_as_int(v));
    else           atomicMin((unsigned int*)a, __float_as_uint(v));
}

__global__ void k_fill(float* p, float v, long n) {
    long i = (long)blockIdx.x * blockDim.x + threadIdx.x;
    if (i < n) p[i] = v;
}

__global__ void k_f32_to_f16(const float* __restrict__ in, _Float16* __restrict__ out, long n) {
    long i = (long)blockIdx.x * blockDim.x + threadIdx.x;
    if (i < n) out[i] = (_Float16)in[i];
}

// BatchNorm1d (eval, mean=0, var=1) -> f16 features for WMMA
__global__ void k_bn_f16(const float* __restrict__ x, const float* __restrict__ g,
                         const float* __restrict__ b, _Float16* __restrict__ out, long n) {
    long i = (long)blockIdx.x * blockDim.x + threadIdx.x;
    if (i >= n) return;
    int c = (int)(i & (FDIM - 1));
    float rs = rsqrtf(1.0f + LN_EPS);
    out[i] = (_Float16)(x[i] * rs * g[c] + b[c]);
}

// ---------------------------------------------------------------- WMMA GEMM
// out[N,64] = A[N,K] (f16) @ W[K,64] (f16), f32 accumulate.
// grid.x = ceil(N/16); block = 128 threads = 4 waves; wave w owns columns
// [16w, 16w+16). W is staged coalesced into LDS, then swizzled once into
// per-(kchunk, coltile, lane) contiguous 16-half B-fragments so the steady
// state inner loop is 2x ds_load_b128 + 2x global_load_b128 + 1x v_wmma.
// A/B fragments use contiguous k = 16*(lane/16)+s, C/D per CDNA5 ISA §7.12.2 (16-bit A 16x32, B 32x16,
// f32 C/D 16x16).
__global__ void k_gemm_wmma(const _Float16* __restrict__ A, const _Float16* __restrict__ W,
                            float* __restrict__ out, int K, int N) {
    __shared__ _Float16 sW[FDIM * HC];                     // raw W tile (<=16KB)
    __shared__ _Float16 sFrag[(FDIM / 32) * 4 * 32 * 16];  // swizzled B fragments (<=16KB)
    int tid = threadIdx.x;
    int tot = K * HC;
    for (int i = tid; i < tot; i += 128) sW[i] = W[i];
    __syncthreads();

    // One-time swizzle: fragment (kc, ct, l) = 16 contiguous halves.
    // B layout (contiguous k, same as A): lanes 0-15 hold K = kc*32+0..15, lanes 16-31 hold
    // K = kc*32+16..31, column N = ct*16 + (l&15).
    int nfrag = (K >> 5) * 4 * 32;
    for (int idx = tid; idx < nfrag; idx += 128) {
        int l  = idx & 31;
        int ct = (idx >> 5) & 3;
        int kc = idx >> 7;
        int kb = (l & 16) ? 16 : 0;
        int c  = (ct << 4) + (l & 15);
        const _Float16* wp = &sW[(kc * 32 + kb) * HC + c];
        _Float16* fp = &sFrag[idx << 4];
#pragma unroll
        for (int i = 0; i < 16; ++i) fp[i] = wp[i * HC];
    }
    __syncthreads();

    int wave = tid >> 5, lane = tid & 31;
    int row0 = blockIdx.x << 4;
    int col  = (wave << 4) + (lane & 15);
    int arow = row0 + (lane & 15);
    if (arow >= N) arow = N - 1;                // clamp (reads only)
    const _Float16* Abase = A + (long)arow * K;
    int kbA = (lane & 16) ? 16 : 0;             // A: lanes 0-15 K0..15, 16-31 K16..31
    const _Float16* fbase = &sFrag[((wave << 5) + lane) << 4];

    v8f acc = {};
    int nkc = K >> 5;
    for (int kc = 0; kc < nkc; ++kc) {
        int k0 = kc << 5;
        if (k0 + 32 < K) __builtin_prefetch(Abase + k0 + 32, 0, 1);
        v16h a;
        const _Float16* ap = Abase + k0 + kbA;
#pragma unroll
        for (int i = 0; i < 8; ++i) { a[i] = ap[i]; a[8 + i] = ap[8 + i]; }
        v16h b = *(const v16h*)(fbase + (kc << 11));   // kc * (4*32*16)
        acc = __builtin_amdgcn_wmma_f32_16x16x32_f16(
            /*neg_a=*/false, a, /*neg_b=*/false, b,
            /*c_mod=*/(short)0, acc, /*reuse_a=*/false, /*reuse_b=*/false);
    }
    int m0 = row0 + ((lane & 16) ? 8 : 0);
#pragma unroll
    for (int j = 0; j < 8; ++j) {
        int r = m0 + j;
        if (r < N) out[(long)r * HC + col] = acc[j];
    }
}

// ---------------------------------------------------------------- attention
// s_src[n,h] = sum_c h[n,h,c]*a_s[h,c];  s_dst analogous
__global__ void k_scores(const float* __restrict__ h, const float* __restrict__ as_,
                         const float* __restrict__ ad_, float* __restrict__ ssrc,
                         float* __restrict__ sdst, int N) {
    int t = blockIdx.x * blockDim.x + threadIdx.x;
    if (t >= N * NHEAD) return;
    int n = t >> 1, hh = t & 1;
    const float* hp = h + (long)n * HC + hh * CPH;
    const float* ap = as_ + hh * CPH;
    const float* dp = ad_ + hh * CPH;
    float ss = 0.f, sd = 0.f;
#pragma unroll 8
    for (int c = 0; c < CPH; ++c) { float v = hp[c]; ss += v * ap[c]; sd += v * dp[c]; }
    ssrc[t] = ss; sdst[t] = sd;
}

// edges: e < E from the edge list, e >= E are self-loops (node e-E)
__global__ void k_edge_max(const int* __restrict__ src, const int* __restrict__ dst,
                           const float* __restrict__ ssrc, const float* __restrict__ sdst,
                           float* __restrict__ m, int E, int N) {
    int e = blockIdx.x * blockDim.x + threadIdx.x;
    if (e >= E + N) return;
    int s = (e < E) ? src[e] : (e - E);
    int d = (e < E) ? dst[e] : (e - E);
#pragma unroll
    for (int hh = 0; hh < NHEAD; ++hh) {
        float v = ssrc[s * NHEAD + hh] + sdst[d * NHEAD + hh];
        v = (v > 0.f) ? v : NEG_SLOPE * v;
        atomicMaxF(&m[d * NHEAD + hh], v);
    }
}

__global__ void k_edge_expsum(const int* __restrict__ src, const int* __restrict__ dst,
                              const float* __restrict__ ssrc, const float* __restrict__ sdst,
                              const float* __restrict__ m, float* __restrict__ p,
                              float* __restrict__ z, int E, int N) {
    int e = blockIdx.x * blockDim.x + threadIdx.x;
    if (e >= E + N) return;
    int s = (e < E) ? src[e] : (e - E);
    int d = (e < E) ? dst[e] : (e - E);
#pragma unroll
    for (int hh = 0; hh < NHEAD; ++hh) {
        float v = ssrc[s * NHEAD + hh] + sdst[d * NHEAD + hh];
        v = (v > 0.f) ? v : NEG_SLOPE * v;
        float pv = expf(v - m[d * NHEAD + hh]);
        p[e * NHEAD + hh] = pv;
        atomicAdd(&z[d * NHEAD + hh], pv);
    }
}

// one thread per (edge, channel): agg[dst, c] += h[src, c] * p/z
__global__ void k_edge_agg(const int* __restrict__ src, const int* __restrict__ dst,
                           const float* __restrict__ h, const float* __restrict__ p,
                           const float* __restrict__ z, float* __restrict__ agg,
                           int E, int N) {
    long t = (long)blockIdx.x * blockDim.x + threadIdx.x;
    long tot = (long)(E + N) * HC;
    if (t >= tot) return;
    int e = (int)(t >> 6);
    int c = (int)(t & 63);
    int hh = c >> 5;
    int s = (e < E) ? src[e] : (e - E);
    int d = (e < E) ? dst[e] : (e - E);
    float w = p[e * NHEAD + hh] / z[d * NHEAD + hh];
    atomicAdd(&agg[(long)d * HC + c], h[(long)s * HC + c] * w);
}

// bias + ELU, write JK concat slot and f16 copy for next layer's WMMA
__global__ void k_finish_layer(const float* __restrict__ agg, const float* __restrict__ bias,
                               float* __restrict__ hs, _Float16* __restrict__ h16,
                               int layer, int N) {
    long t = (long)blockIdx.x * blockDim.x + threadIdx.x;
    if (t >= (long)N * HC) return;
    int c = (int)(t & 63);
    long n = t >> 6;
    float v = agg[t] + bias[c];
    v = (v > 0.f) ? v : expm1f(v);              // ELU (alpha=1)
    hs[n * JKDIM + layer * HC + c] = v;
    h16[t] = (_Float16)v;
}

// ---------------------------------------------------------------- pooling
// one block (192 threads) per node: LayerNorm over 192, then segment max/sum
__global__ void k_ln_pool(const float* __restrict__ hs, const float* __restrict__ g,
                          const float* __restrict__ bta, const int* __restrict__ batch,
                          float* __restrict__ pmax, float* __restrict__ psum,
                          float* __restrict__ cnt, int N) {
    __shared__ float sd[JKDIM];
    int n = blockIdx.x;
    int j = threadIdx.x;
    float v = hs[(long)n * JKDIM + j];
    sd[j] = v; __syncthreads();
    for (int st = 96; st >= 3; st >>= 1) { if (j < st) sd[j] += sd[j + st]; __syncthreads(); }
    float mu = (sd[0] + sd[1] + sd[2]) * (1.0f / JKDIM);
    __syncthreads();
    float dv = v - mu;
    sd[j] = dv * dv; __syncthreads();
    for (int st = 96; st >= 3; st >>= 1) { if (j < st) sd[j] += sd[j + st]; __syncthreads(); }
    float var = (sd[0] + sd[1] + sd[2]) * (1.0f / JKDIM);
    float y = dv * rsqrtf(var + LN_EPS) * g[j] + bta[j];
    int bi = batch[n];
    atomicMaxF(&pmax[bi * JKDIM + j], y);
    atomicAdd(&psum[bi * JKDIM + j], y);
    if (j == 0) atomicAdd(&cnt[bi], 1.0f);
}

// hv = [max | mean] (384) ; view = relu(hv @ projW + pb). grid=B, block=128
__global__ void k_pool_proj(const float* __restrict__ pmax, const float* __restrict__ psum,
                            const float* __restrict__ cnt, const float* __restrict__ pW,
                            const float* __restrict__ pb, float* __restrict__ view) {
    __shared__ float shv[2 * JKDIM];
    int b = blockIdx.x, tid = threadIdx.x;
    float inv = 1.0f / fmaxf(cnt[b], 1.0f);
    for (int i = tid; i < 2 * JKDIM; i += 128)
        shv[i] = (i < JKDIM) ? pmax[b * JKDIM + i] : psum[b * JKDIM + (i - JKDIM)] * inv;
    __syncthreads();
    float acc = pb[tid];
    for (int jj = 0; jj < 2 * JKDIM; ++jj) acc += shv[jj] * pW[jj * 128 + tid];
    view[(long)b * 128 + tid] = fmaxf(acc, 0.0f);
}

// ---------------------------------------------------------------- fusion head
// one thread per graph (1 block of 256)
__global__ void k_fuse(const float* __restrict__ views,
                       const float* __restrict__ gW1, const float* __restrict__ gb1,
                       const float* __restrict__ gW2, const float* __restrict__ gb2,
                       const float* __restrict__ cW1, const float* __restrict__ cb1,
                       const float* __restrict__ cW2, const float* __restrict__ cb2,
                       const float* __restrict__ cW3, const float* __restrict__ cb3,
                       float* __restrict__ out) {
    int b = threadIdx.x;
    const float* v0 = views + (long)b * 128;
    const float* v1 = views + (long)NB * 128 + (long)b * 128;
    const float* v2 = views + (long)2 * NB * 128 + (long)b * 128;
    float gi[128];
    for (int c = 0; c < 128; ++c) gi[c] = (v0[c] + v1[c] + v2[c]) * (1.0f / 3.0f);
    float g1[64];
    for (int jj = 0; jj < 64; ++jj) {
        float a = gb1[jj];
        for (int c = 0; c < 128; ++c) a += gi[c] * gW1[c * 64 + jj];
        g1[jj] = fmaxf(a, 0.0f);
    }
    float lg[3]; float mx = -3.0e38f;
    for (int k = 0; k < 3; ++k) {
        float a = gb2[k];
        for (int jj = 0; jj < 64; ++jj) a += g1[jj] * gW2[jj * 3 + k];
        lg[k] = a; mx = fmaxf(mx, a);
    }
    float sum = 0.f;
    for (int k = 0; k < 3; ++k) { lg[k] = expf(lg[k] - mx); sum += lg[k]; }
    float a0 = lg[0] / sum, a1 = lg[1] / sum, a2 = lg[2] / sum;
    float fused[128];
    for (int c = 0; c < 128; ++c) fused[c] = a0 * v0[c] + a1 * v1[c] + a2 * v2[c];
    float h1[128];
    for (int jj = 0; jj < 128; ++jj) {
        float a = cb1[jj];
        for (int c = 0; c < 128; ++c) a += fused[c] * cW1[c * 128 + jj];
        h1[jj] = fmaxf(a, 0.0f);
    }
    float h2[64];
    for (int jj = 0; jj < 64; ++jj) {
        float a = cb2[jj];
        for (int c = 0; c < 128; ++c) a += h1[c] * cW2[c * 64 + jj];
        h2[jj] = fmaxf(a, 0.0f);
    }
    float o = cb3[0];
    for (int jj = 0; jj < 64; ++jj) o += h2[jj] * cW3[jj];
    out[b] = o;
}

// ---------------------------------------------------------------- launch

extern "C" void kernel_launch(void* const* d_in, const int* in_sizes, int n_in,
                              void* d_out, int out_size, void* d_ws, size_t ws_size,
                              hipStream_t stream) {
    (void)n_in; (void)out_size; (void)ws_size;
    const float* x      = (const float*)d_in[0];
    const int*   edges[3] = {(const int*)d_in[1], (const int*)d_in[2], (const int*)d_in[3]};
    const int*   batch  = (const int*)d_in[4];
    const float* bn_g   = (const float*)d_in[5];
    const float* bn_b   = (const float*)d_in[6];
    const float* Wl[3]  = {(const float*)d_in[7],  (const float*)d_in[11], (const float*)d_in[15]};
    const float* asl[3] = {(const float*)d_in[8],  (const float*)d_in[12], (const float*)d_in[16]};
    const float* adl[3] = {(const float*)d_in[9],  (const float*)d_in[13], (const float*)d_in[17]};
    const float* bl[3]  = {(const float*)d_in[10], (const float*)d_in[14], (const float*)d_in[18]};
    const float* ln_g   = (const float*)d_in[19];
    const float* ln_b   = (const float*)d_in[20];
    const float* pW     = (const float*)d_in[21];
    const float* pb     = (const float*)d_in[22];
    const float* gW1    = (const float*)d_in[23];
    const float* gb1    = (const float*)d_in[24];
    const float* gW2    = (const float*)d_in[25];
    const float* gb2    = (const float*)d_in[26];
    const float* cW1    = (const float*)d_in[27];
    const float* cb1    = (const float*)d_in[28];
    const float* cW2    = (const float*)d_in[29];
    const float* cb2    = (const float*)d_in[30];
    const float* cW3    = (const float*)d_in[31];
    const float* cb3    = (const float*)d_in[32];
    float* out = (float*)d_out;

    const int N  = in_sizes[4];
    const int E  = in_sizes[1] / 2;
    const int Ep = E + N;

    // --- carve workspace (256B aligned slices)
    size_t off = 0;
    auto carve = [&](size_t bytes) -> void* {
        void* ptr = (char*)d_ws + off;
        off += (bytes + 255) & ~(size_t)255;
        return ptr;
    };
    _Float16* xn16 = (_Float16*)carve((size_t)N * FDIM * 2);
    _Float16* h16  = (_Float16*)carve((size_t)N * HC * 2);
    _Float16* w16  = (_Float16*)carve((size_t)FDIM * HC * 2);
    float* hpre = (float*)carve((size_t)N * HC * 4);
    float* ssrc = (float*)carve((size_t)N * NHEAD * 4);
    float* sdst = (float*)carve((size_t)N * NHEAD * 4);
    float* mbuf = (float*)carve((size_t)N * NHEAD * 4);
    float* zbuf = (float*)carve((size_t)N * NHEAD * 4);
    float* pbuf = (float*)carve((size_t)Ep * NHEAD * 4);
    float* agg  = (float*)carve((size_t)N * HC * 4);
    float* hs   = (float*)carve((size_t)N * JKDIM * 4);
    float* pmax = (float*)carve((size_t)NB * JKDIM * 4);
    float* psum = (float*)carve((size_t)NB * JKDIM * 4);
    float* cnt  = (float*)carve((size_t)NB * 4);
    float* views = (float*)carve((size_t)3 * NB * 128 * 4);

    auto cdiv = [](long a, long b) -> int { return (int)((a + b - 1) / b); };

    k_bn_f16<<<cdiv((long)N * FDIM, 256), 256, 0, stream>>>(x, bn_g, bn_b, xn16, (long)N * FDIM);

    const int Kin[3] = {FDIM, HC, HC};
    for (int v = 0; v < 3; ++v) {
        const int* src = edges[v];
        const int* dst = edges[v] + E;
        for (int l = 0; l < 3; ++l) {
            const int K = Kin[l];
            k_f32_to_f16<<<cdiv((long)K * HC, 256), 256, 0, stream>>>(
                Wl[l] + (size_t)v * K * HC, w16, (long)K * HC);
            const _Float16* Ain = (l == 0) ? xn16 : h16;
            k_gemm_wmma<<<cdiv(N, 16), 128, 0, stream>>>(Ain, w16, hpre, K, N);
            k_scores<<<cdiv((long)N * NHEAD, 256), 256, 0, stream>>>(
                hpre, asl[l] + (size_t)v * HC, adl[l] + (size_t)v * HC, ssrc, sdst, N);
            k_fill<<<cdiv((long)N * NHEAD, 256), 256, 0, stream>>>(mbuf, -3.0e38f, (long)N * NHEAD);
            k_fill<<<cdiv((long)N * NHEAD, 256), 256, 0, stream>>>(zbuf, 0.0f, (long)N * NHEAD);
            k_edge_max<<<cdiv(Ep, 256), 256, 0, stream>>>(src, dst, ssrc, sdst, mbuf, E, N);
            k_edge_expsum<<<cdiv(Ep, 256), 256, 0, stream>>>(src, dst, ssrc, sdst, mbuf, pbuf, zbuf, E, N);
            k_fill<<<cdiv((long)N * HC, 256), 256, 0, stream>>>(agg, 0.0f, (long)N * HC);
            k_edge_agg<<<cdiv((long)Ep * HC, 256), 256, 0, stream>>>(src, dst, hpre, pbuf, zbuf, agg, E, N);
            k_finish_layer<<<cdiv((long)N * HC, 256), 256, 0, stream>>>(
                agg, bl[l] + (size_t)v * HC, hs, h16, l, N);
        }
        k_fill<<<cdiv((long)NB * JKDIM, 256), 256, 0, stream>>>(pmax, -3.0e38f, (long)NB * JKDIM);
        k_fill<<<cdiv((long)NB * JKDIM, 256), 256, 0, stream>>>(psum, 0.0f, (long)NB * JKDIM);
        k_fill<<<1, 256, 0, stream>>>(cnt, 0.0f, NB);
        k_ln_pool<<<N, JKDIM, 0, stream>>>(hs, ln_g, ln_b, batch, pmax, psum, cnt, N);
        k_pool_proj<<<NB, 128, 0, stream>>>(pmax, psum, cnt, pW, pb, views + (size_t)v * NB * 128);
    }
    k_fuse<<<1, NB, 0, stream>>>(views, gW1, gb1, gW2, gb2, cW1, cb1, cW2, cb2, cW3, cb3, out);
}
